// SREL_54820962566423
// MI455X (gfx1250) — compile-verified, matrix operand order checked by reading, not run on hardware
//
#include <hip/hip_runtime.h>
#include <hip/hip_bf16.h>

typedef __attribute__((ext_vector_type(16))) _Float16 v16h;
typedef __attribute__((ext_vector_type(8)))  float    v8f;

#define BB 256
#define LS 1024
#define LW 1024
#define MMD 16
#define DD 5120
#define HH 2048
#define MT (BB*MMD)   // 4096 token rows

// ---------------------------------------------------------------------------
// Build X = [mag*cos(phi) | mag*sin(phi) | w_real^T | w_imag^T | v^T] in f16
// ---------------------------------------------------------------------------
__global__ __launch_bounds__(256)
void build_x_kernel(const float* __restrict__ phi,
                    const float* __restrict__ wr,
                    const float* __restrict__ wi,
                    const float* __restrict__ vM,
                    _Float16* __restrict__ X)
{
    size_t idx = (size_t)blockIdx.x * 256 + threadIdx.x;
    if (idx >= (size_t)MT * DD) return;
    int d   = (int)(idx % DD);
    int row = (int)(idx / DD);
    int b = row >> 4, m = row & 15;
    const float mag = 0.03125f;   // 1/sqrt(1024)
    float val;
    if (d < LS)                 val = mag * cosf(phi[(size_t)b*LS + d]);
    else if (d < 2*LS)          val = mag * sinf(phi[(size_t)b*LS + (d - LS)]);
    else if (d < 2*LS + LW)     val = wr[((size_t)b*LW + (d - 2*LS)) * MMD + m];
    else if (d < 2*LS + 2*LW)   val = wi[((size_t)b*LW + (d - 2*LS - LW)) * MMD + m];
    else                        val = vM[(size_t)(d - (2*LS + 2*LW)) * MMD + m];
    X[idx] = (_Float16)val;
}

// ---------------------------------------------------------------------------
// Transpose-convert: in[K][N] fp32  ->  out[N][K] f16 (so GEMM B-tiles are
// k-contiguous and need no in-loop LDS transpose).
// ---------------------------------------------------------------------------
__global__ __launch_bounds__(256)
void transpose_f32_to_f16_kernel(const float* __restrict__ in,
                                 _Float16* __restrict__ out,
                                 int K, int N)
{
    __shared__ float tile[32][33];
    const int k0 = blockIdx.y * 32;
    const int n0 = blockIdx.x * 32;
    const int tx = threadIdx.x & 31;
    const int ty = threadIdx.x >> 5;     // 0..7
    #pragma unroll
    for (int i = ty; i < 32; i += 8)
        tile[i][tx] = in[(size_t)(k0 + i) * N + n0 + tx];
    __syncthreads();
    #pragma unroll
    for (int i = ty; i < 32; i += 8)
        out[(size_t)(n0 + i) * K + k0 + tx] = (_Float16)tile[tx][i];
}

// ---------------------------------------------------------------------------
// Tiled WMMA GEMM: C[M x N] = op(A[M x K] @ Bt[N x K]^T + bias)
// Block tile 128x128, K-step 32, double-buffered LDS (one barrier/iter).
// 8 wave32 waves: 4x2 grid, each wave 32x64 = 2x4 v_wmma_f32_16x16x32_f16.
// ---------------------------------------------------------------------------
template<bool RELU, bool OUT_F16>
__global__ __launch_bounds__(256)
void gemm_wmma_kernel(const _Float16* __restrict__ A,    // [M][K]
                      const _Float16* __restrict__ Bt,   // [N][K]
                      const float* __restrict__ bias,    // [N]
                      void* __restrict__ Cout,           // [M][N]
                      int N, int K)
{
    __shared__ _Float16 As[2][128][40];   // [buf][m][k]
    __shared__ _Float16 Bs[2][128][40];   // [buf][n][k]

    const int t    = threadIdx.x;
    const int lane = t & 31;
    const int wid  = t >> 5;          // 0..7
    const int wm   = wid & 3;         // 4 wave-rows * 32 rows
    const int wn   = wid >> 2;        // 2 wave-cols * 64 cols
    const int half = lane >> 4;       // K-half selector (ISA f16 layout)
    const int l15  = lane & 15;

    const int row0 = blockIdx.y * 128;
    const int col0 = blockIdx.x * 128;

    // global->LDS copy mapping: 256 threads, each owns one 16-f16 chunk of
    // the 128x32 A tile and one of the 128x32 Bt tile (both k-contiguous).
    const int lr = t >> 1;            // tile row 0..127
    const int lk = (t & 1) * 16;      // k-chunk 0 / 16

    const _Float16* aP = A  + (size_t)(row0 + lr) * K + lk;
    const _Float16* bP = Bt + (size_t)(col0 + lr) * K + lk;

    v8f zero = {};
    v8f acc[2][4];
    #pragma unroll
    for (int i = 0; i < 2; ++i)
        #pragma unroll
        for (int j = 0; j < 4; ++j) acc[i][j] = zero;

    { // prologue: stage K-tile 0 into buffer 0
        uint4 a0 = *reinterpret_cast<const uint4*>(aP);
        uint4 a1 = *reinterpret_cast<const uint4*>(aP + 8);
        uint4 b0 = *reinterpret_cast<const uint4*>(bP);
        uint4 b1 = *reinterpret_cast<const uint4*>(bP + 8);
        *reinterpret_cast<uint4*>(&As[0][lr][lk])     = a0;
        *reinterpret_cast<uint4*>(&As[0][lr][lk + 8]) = a1;
        *reinterpret_cast<uint4*>(&Bs[0][lr][lk])     = b0;
        *reinterpret_cast<uint4*>(&Bs[0][lr][lk + 8]) = b1;
    }
    __syncthreads();

    const int nkt = K >> 5;
    for (int kt = 0; kt < nkt; ++kt) {
        const int  cur  = kt & 1;
        const int  nxt  = cur ^ 1;
        const bool more = (kt + 1) < nkt;

        // issue next tile's global loads early (overlap with WMMA below)
        uint4 a0, a1, b0, b1;
        if (more) {
            const _Float16* ap = aP + ((size_t)(kt + 1) << 5);
            const _Float16* bp = bP + ((size_t)(kt + 1) << 5);
            a0 = *reinterpret_cast<const uint4*>(ap);
            a1 = *reinterpret_cast<const uint4*>(ap + 8);
            b0 = *reinterpret_cast<const uint4*>(bp);
            b1 = *reinterpret_cast<const uint4*>(bp + 8);
            if (kt + 2 < nkt) { // pull K-tile kt+2 toward L2 (global_prefetch_b8)
                __builtin_prefetch(aP + ((size_t)(kt + 2) << 5), 0, 1);
                __builtin_prefetch(bP + ((size_t)(kt + 2) << 5), 0, 1);
            }
        }

        // fragment builds per ISA 16-bit layout: lane-half h takes K 8h..8h+7 and 16+8h..16+8h+7
        union { uint4 u[2]; v16h v; } af[2], bf[4];
        #pragma unroll
        for (int mi = 0; mi < 2; ++mi) {
            const int r = wm * 32 + mi * 16 + l15;
            af[mi].u[0] = *reinterpret_cast<const uint4*>(&As[cur][r][half * 8]);
            af[mi].u[1] = *reinterpret_cast<const uint4*>(&As[cur][r][16 + half * 8]);
        }
        #pragma unroll
        for (int ni = 0; ni < 4; ++ni) {
            const int c = wn * 64 + ni * 16 + l15;
            bf[ni].u[0] = *reinterpret_cast<const uint4*>(&Bs[cur][c][half * 8]);
            bf[ni].u[1] = *reinterpret_cast<const uint4*>(&Bs[cur][c][16 + half * 8]);
        }
        #pragma unroll
        for (int mi = 0; mi < 2; ++mi)
            #pragma unroll
            for (int ni = 0; ni < 4; ++ni)
                acc[mi][ni] = __builtin_amdgcn_wmma_f32_16x16x32_f16(
                    false, af[mi].v, false, bf[ni].v,
                    (short)0, acc[mi][ni], false, false);

        // stage next tile into the other buffer (no conflict with cur reads)
        if (more) {
            *reinterpret_cast<uint4*>(&As[nxt][lr][lk])     = a0;
            *reinterpret_cast<uint4*>(&As[nxt][lr][lk + 8]) = a1;
            *reinterpret_cast<uint4*>(&Bs[nxt][lr][lk])     = b0;
            *reinterpret_cast<uint4*>(&Bs[nxt][lr][lk + 8]) = b1;
        }
        __syncthreads();
    }

    // Epilogue: bias (+ReLU) and store. C layout: VGPR r -> M = r + 8*half.
    #pragma unroll
    for (int mi = 0; mi < 2; ++mi) {
        #pragma unroll
        for (int ni = 0; ni < 4; ++ni) {
            const int n  = col0 + wn * 64 + ni * 16 + l15;
            const float bv = bias[n];
            #pragma unroll
            for (int r = 0; r < 8; ++r) {
                const int m = row0 + wm * 32 + mi * 16 + half * 8 + r;
                float v = acc[mi][ni][r] + bv;
                if (RELU) v = v > 0.0f ? v : 0.0f;
                if (OUT_F16)
                    reinterpret_cast<_Float16*>(Cout)[(size_t)m * N + n] = (_Float16)v;
                else
                    reinterpret_cast<float*>(Cout)[(size_t)m * N + n] = v;
            }
        }
    }
}

// ---------------------------------------------------------------------------
// rho[row] = H1r[row,:] . W2r + b2r   (4096 rows, K = 2048)
// ---------------------------------------------------------------------------
__global__ __launch_bounds__(256)
void rho_kernel(const _Float16* __restrict__ Hr, const float* __restrict__ W2r,
                const float* __restrict__ b2r, float* __restrict__ rho)
{
    __shared__ float red[256];
    const int row = blockIdx.x;
    float s = 0.0f;
    for (int i = threadIdx.x; i < HH; i += 256)
        s += (float)Hr[(size_t)row * HH + i] * W2r[i];
    red[threadIdx.x] = s;
    __syncthreads();
    for (int off = 128; off > 0; off >>= 1) {
        if (threadIdx.x < off) red[threadIdx.x] += red[threadIdx.x + off];
        __syncthreads();
    }
    if (threadIdx.x == 0) rho[row] = red[0] + b2r[0];
}

// ---------------------------------------------------------------------------
// phi_opt[b,l] = phi[b,l] - sum_m rho[b,m]*EtaM[b,m,l];  also write EtaM^T
// ---------------------------------------------------------------------------
__global__ __launch_bounds__(256)
void final_kernel(const float* __restrict__ phi,
                  const float* __restrict__ EtaM,
                  const float* __restrict__ rho,
                  float* __restrict__ out_phi,
                  float* __restrict__ out_etaMT)
{
    int idx = blockIdx.x * 256 + threadIdx.x;     // b*LS + l
    int b = idx >> 10, l = idx & (LS - 1);
    float eta = 0.0f;
    #pragma unroll
    for (int m = 0; m < MMD; ++m) {
        float v = EtaM[((size_t)(b * MMD + m)) * LS + l];
        eta += rho[b * MMD + m] * v;
        out_etaMT[((size_t)b * LS + l) * MMD + m] = v;  // [B, LS, M]
    }
    out_phi[idx] = phi[idx] - eta;
}

// ---------------------------------------------------------------------------
extern "C" void kernel_launch(void* const* d_in, const int* in_sizes, int n_in,
                              void* d_out, int out_size, void* d_ws, size_t ws_size,
                              hipStream_t stream)
{
    (void)in_sizes; (void)n_in; (void)out_size; (void)ws_size;
    const float* phi = (const float*)d_in[0];
    const float* wr  = (const float*)d_in[1];
    const float* wi  = (const float*)d_in[2];
    const float* vM  = (const float*)d_in[3];
    const float* W1e = (const float*)d_in[4];
    const float* b1e = (const float*)d_in[5];
    const float* W2e = (const float*)d_in[6];
    const float* b2e = (const float*)d_in[7];
    const float* W1r = (const float*)d_in[8];
    const float* b1r = (const float*)d_in[9];
    const float* W2r = (const float*)d_in[10];
    const float* b2r = (const float*)d_in[11];

    char* ws = (char*)d_ws;
    size_t off = 0;
    _Float16* Xh    = (_Float16*)(ws + off); off += (size_t)MT * DD * 2;
    _Float16* W1eT  = (_Float16*)(ws + off); off += (size_t)DD * HH * 2;  // [H][D]
    _Float16* W1rT  = (_Float16*)(ws + off); off += (size_t)DD * HH * 2;  // [H][D]
    _Float16* W2eT  = (_Float16*)(ws + off); off += (size_t)HH * LS * 2;  // [LS][H]
    _Float16* H1e   = (_Float16*)(ws + off); off += (size_t)MT * HH * 2;
    _Float16* H1r   = (_Float16*)(ws + off); off += (size_t)MT * HH * 2;
    float*    EtaM  = (float*)(ws + off);    off += (size_t)MT * LS * 4;
    float*    rho   = (float*)(ws + off);    off += (size_t)MT * 4;

    { // featurize + concat, fp32 -> f16
        size_t n = (size_t)MT * DD;
        build_x_kernel<<<(unsigned)((n + 255) / 256), 256, 0, stream>>>(phi, wr, wi, vM, Xh);
    }
    // weight transpose-conversions: [K][N] fp32 -> [N][K] f16
    transpose_f32_to_f16_kernel<<<dim3(HH / 32, DD / 32), 256, 0, stream>>>(W1e, W1eT, DD, HH);
    transpose_f32_to_f16_kernel<<<dim3(HH / 32, DD / 32), 256, 0, stream>>>(W1r, W1rT, DD, HH);
    transpose_f32_to_f16_kernel<<<dim3(LS / 32, HH / 32), 256, 0, stream>>>(W2e, W2eT, HH, LS);

    // Layer 1 (both heads): [4096x5120] @ [5120x2048], ReLU, f16 out
    gemm_wmma_kernel<true, true><<<dim3(HH / 128, MT / 128), 256, 0, stream>>>(
        Xh, W1eT, b1e, (void*)H1e, HH, DD);
    gemm_wmma_kernel<true, true><<<dim3(HH / 128, MT / 128), 256, 0, stream>>>(
        Xh, W1rT, b1r, (void*)H1r, HH, DD);
    // Layer 2 (eta head): [4096x2048] @ [2048x1024], fp32 out
    gemm_wmma_kernel<false, false><<<dim3(LS / 128, MT / 128), 256, 0, stream>>>(
        H1e, W2eT, b2e, (void*)EtaM, LS, HH);
    // rho head (N = 1): plain reduction
    rho_kernel<<<MT, 256, 0, stream>>>(H1r, W2r, b2r, rho);

    float* out_phi   = (float*)d_out;
    float* out_etaMT = out_phi + (size_t)BB * LS;
    final_kernel<<<(BB * LS) / 256, 256, 0, stream>>>(phi, EtaM, rho, out_phi, out_etaMT);
}
